// CrossScaleFreqAttention_24378234372601
// MI455X (gfx1250) — matrix operand order repaired and checked by reading, not compiled, not run on hardware
//
#include <hip/hip_runtime.h>

typedef __attribute__((ext_vector_type(16))) _Float16 v16h;
typedef __attribute__((ext_vector_type(8)))  float    v8f;

#define BB   8
#define LL   1024
#define CCH  64
#define KNB  4
#define DDIM 32
#define NTOK 5
#define WIN  64            // W*W
#define CHUNK (CCH * WIN)  // 4096 floats per (b,l)

__device__ __forceinline__ void lds_fence() {
  // intra-wave cross-lane LDS RAW: hardware keeps DS ops in order per wave,
  // this forces the dscnt wait + acts as a compiler memory fence.
  asm volatile("s_wait_dscnt 0" ::: "memory");
}

// A fragment (f16, 16xK slice of 32 from an f32 LDS matrix, row stride rs)
// lanes 0-15: M=lane, h[0..7]=K0..7,  h[8..15]=K16..23
// lanes16-31: M=lane-16, h[0..7]=K8..15, h[8..15]=K24..31
__device__ __forceinline__ v16h load_a_frag_f32(const float* m, int rs, int k0, int lane) {
  const int row = lane & 15;
  const int hb  = (lane >> 4) * 8;
  const float* r0 = m + row * rs + k0;
  v16h a;
#pragma unroll
  for (int i = 0; i < 8; ++i) {
    a[i]     = (_Float16)r0[hb + i];
    a[i + 8] = (_Float16)r0[16 + hb + i];
  }
  return a;
}

// B fragment (f16, 32x16 slice of weight matrix w[Kdim][Ndim], row stride ns)
// lanes 0-15: N=lane, h[0..7]=K0..7, h[8..15]=K16..23; lanes 16-31: N=lane-16, h[0..7]=K8..15, h[8..15]=K24..31
__device__ __forceinline__ v16h load_b_frag(const float* w, int ns, int k0, int n0, int lane) {
  const int col = (lane & 15) + n0;
  const int kb  = k0 + (lane >> 4) * 8;
  v16h b;
#pragma unroll
  for (int i = 0; i < 16; ++i)
    b[i] = (_Float16)w[(kb + i + (i >= 8 ? 8 : 0)) * ns + col];
  return b;
}

__global__ __launch_bounds__(256)
void CrossScaleFreqAttention_kernel(
    const float* __restrict__ target, const float* __restrict__ neigh,
    const float* __restrict__ proj_w, const float* __restrict__ proj_b,
    const float* __restrict__ q_w,  const float* __restrict__ q_b,
    const float* __restrict__ k_w,  const float* __restrict__ k_b,
    const float* __restrict__ v_w,  const float* __restrict__ v_b,
    const float* __restrict__ out_w, const float* __restrict__ out_b,
    float* __restrict__ outp) {
  __shared__ float tokLDS[16 * 64];        // tokens, rows 5..15 zero
  __shared__ float projLDS[16 * 33];       // padded stride 33
  __shared__ float kmatLDS[16 * 33];
  __shared__ float vmatLDS[16 * 33];
  __shared__ float qLDS[32];
  __shared__ float fusedLDS[32];
  __shared__ float deltaLDS[64];

  const int tid  = threadIdx.x;
  const int wv   = tid >> 5;
  const int lane = tid & 31;
  const int g    = blockIdx.x;             // flat (b*L + l), 0..8191
  const float* tgt = target + (size_t)g * CHUNK;

  // ---------------- Phase 0: zero token pad rows ----------------
  for (int i = tid; i < 16 * 64; i += 256) tokLDS[i] = 0.f;
  __syncthreads();

  // ---------------- Phase 1: mean-pool 5 window stacks ----------------
  // wave w handles channels [8w, 8w+8); 32 lanes read the 64 window elems.
  for (int t = 0; t < NTOK; ++t) {
    const float* src = (t == 0) ? tgt
                                : (neigh + ((size_t)(t - 1) * (BB * LL) + g) * CHUNK);
#pragma unroll
    for (int cc = 0; cc < 8; ++cc) {
      const int c = wv * 8 + cc;
      float a;
      if (t == 0) {
        a = src[c * WIN + lane] + src[c * WIN + lane + 32];
      } else {
        a = __builtin_nontemporal_load(&src[c * WIN + lane]) +
            __builtin_nontemporal_load(&src[c * WIN + lane + 32]);
      }
#pragma unroll
      for (int off = 16; off; off >>= 1) a += __shfl_xor(a, off, 32);
      if (lane == 0) tokLDS[t * 64 + c] = a * (1.f / 64.f);
    }
  }
  __syncthreads();

  // ---------------- Phase 2: attention math on wave 0 (WMMA) ----------------
  if (tid < 32) {
    // tokens(16x64) @ proj_w(64x32) + proj_b
    v16h a0 = load_a_frag_f32(tokLDS, 64, 0,  lane);
    v16h a1 = load_a_frag_f32(tokLDS, 64, 32, lane);
#pragma unroll
    for (int n0 = 0; n0 < DDIM; n0 += 16) {
      v16h b0 = load_b_frag(proj_w, DDIM, 0,  n0, lane);
      v16h b1 = load_b_frag(proj_w, DDIM, 32, n0, lane);
      v8f acc = {};
      acc = __builtin_amdgcn_wmma_f32_16x16x32_f16(false, a0, false, b0, (short)0, acc, false, false);
      acc = __builtin_amdgcn_wmma_f32_16x16x32_f16(false, a1, false, b1, (short)0, acc, false, false);
      const float bias = proj_b[n0 + (lane & 15)];
      const int   n    = n0 + (lane & 15);
      const int   mrow = (lane >> 4) * 8;
#pragma unroll
      for (int r = 0; r < 8; ++r)
        projLDS[(mrow + r) * 33 + n] = acc[r] + bias;
    }
    lds_fence();

    // Q/K/V = proj(16x32) @ {q,k,v}_w(32x32) + bias
    v16h ap = load_a_frag_f32(projLDS, 33, 0, lane);
#pragma unroll
    for (int n0 = 0; n0 < DDIM; n0 += 16) {
      v16h bq = load_b_frag(q_w, DDIM, 0, n0, lane);
      v16h bk = load_b_frag(k_w, DDIM, 0, n0, lane);
      v16h bv = load_b_frag(v_w, DDIM, 0, n0, lane);
      v8f z = {};
      v8f qa = __builtin_amdgcn_wmma_f32_16x16x32_f16(false, ap, false, bq, (short)0, z, false, false);
      v8f ka = __builtin_amdgcn_wmma_f32_16x16x32_f16(false, ap, false, bk, (short)0, z, false, false);
      v8f va = __builtin_amdgcn_wmma_f32_16x16x32_f16(false, ap, false, bv, (short)0, z, false, false);
      const int n = n0 + (lane & 15);
      if (lane < 16) {                 // rows 0..4 live in VGPRs 0..4 of lanes 0-15
        qLDS[n] = qa[0] + q_b[n];
#pragma unroll
        for (int r = 0; r < NTOK; ++r) {
          kmatLDS[r * 33 + n] = ka[r] + k_b[n];
          vmatLDS[r * 33 + n] = va[r] + v_b[n];
        }
      }
    }
    lds_fence();

    // softmax over N=5 (redundant per lane; trivial), fused[d]=Σ attn·v  (d = lane)
    float s[NTOK];
    float mx = -3.0e38f;
#pragma unroll
    for (int t = 0; t < NTOK; ++t) {
      float acc = 0.f;
#pragma unroll
      for (int d = 0; d < DDIM; ++d) acc += qLDS[d] * kmatLDS[t * 33 + d];
      s[t] = acc * 0.17677669529663687f;   // 1/sqrt(32)
      mx = fmaxf(mx, s[t]);
    }
    float denom = 0.f;
#pragma unroll
    for (int t = 0; t < NTOK; ++t) { s[t] = __expf(s[t] - mx); denom += s[t]; }
    const float inv = 1.f / denom;
    float f = 0.f;
#pragma unroll
    for (int t = 0; t < NTOK; ++t) f += s[t] * vmatLDS[t * 33 + lane];
    fusedLDS[lane] = f * inv;
    lds_fence();

    // delta(64) = fused(32) @ out_w(32x64) + out_b ; lane does c=lane, lane+32
    float acc0 = out_b[lane];
    float acc1 = out_b[lane + 32];
#pragma unroll
    for (int d = 0; d < DDIM; ++d) {
      const float fd = fusedLDS[d];
      acc0 += fd * out_w[d * CCH + lane];
      acc1 += fd * out_w[d * CCH + lane + 32];
    }
    deltaLDS[lane]      = acc0;
    deltaLDS[lane + 32] = acc1;
  }
  __syncthreads();

  // ---------------- Phase 3: out = target + delta[c] (streamed) ----------------
  float* dst = outp + (size_t)g * CHUNK;
#pragma unroll
  for (int i = 0; i < CHUNK / 256; ++i) {
    const int e = i * 256 + tid;
    __builtin_nontemporal_store(tgt[e] + deltaLDS[e >> 6], &dst[e]);
  }
}

extern "C" void kernel_launch(void* const* d_in, const int* in_sizes, int n_in,
                              void* d_out, int out_size, void* d_ws, size_t ws_size,
                              hipStream_t stream) {
  const float* target = (const float*)d_in[0];
  const float* neigh  = (const float*)d_in[1];
  const float* proj_w = (const float*)d_in[2];
  const float* proj_b = (const float*)d_in[3];
  const float* q_w    = (const float*)d_in[4];
  const float* q_b    = (const float*)d_in[5];
  const float* k_w    = (const float*)d_in[6];
  const float* k_b    = (const float*)d_in[7];
  const float* v_w    = (const float*)d_in[8];
  const float* v_b    = (const float*)d_in[9];
  const float* out_w  = (const float*)d_in[10];
  const float* out_b  = (const float*)d_in[11];
  float* outp = (float*)d_out;

  dim3 grid(BB * LL);
  dim3 block(256);
  hipLaunchKernelGGL(CrossScaleFreqAttention_kernel, grid, block, 0, stream,
                     target, neigh, proj_w, proj_b, q_w, q_b, k_w, k_b,
                     v_w, v_b, out_w, out_b, outp);
}
